// Model_4483945857103
// MI455X (gfx1250) — compile-verified, matrix operand order checked by reading, not run on hardware
//
#include <hip/hip_runtime.h>
#include <math.h>

typedef __attribute__((ext_vector_type(16))) __bf16 v16bf;
typedef __attribute__((ext_vector_type(8)))  __bf16 v8bf;
typedef __attribute__((ext_vector_type(8)))  float  v8f;
typedef __attribute__((ext_vector_type(4)))  int    v4i;
typedef __attribute__((address_space(3)))    v4i    lds_v4i;

constexpr int Bb = 256;   // batch
constexpr int Tt = 128;   // time
constexpr int Dd = 128;   // input dim
constexpr int Hh = 1024;  // hidden
constexpr int TT = Tt - 1;
constexpr int G4H = 4 * Hh;

#if __has_builtin(__builtin_amdgcn_global_load_async_to_lds_b128)
#define ASYNC_LDS 1
#else
#define ASYNC_LDS 0
#endif

__device__ __forceinline__ __bf16 f2bf(float f) {
    unsigned u = __float_as_uint(f);
    u += 0x7FFFu + ((u >> 16) & 1u);          // round-to-nearest-even
    unsigned short s = (unsigned short)(u >> 16);
    return *reinterpret_cast<__bf16*>(&s);
}

// 16B global -> LDS copy: CDNA5 async-to-LDS when available, else via VGPRs.
__device__ __forceinline__ void copy16(const __bf16* g, __bf16* l) {
#if ASYNC_LDS
    __builtin_amdgcn_global_load_async_to_lds_b128(
        (v4i*)(g), (lds_v4i*)(l), 0, 0);
#else
    *reinterpret_cast<v8bf*>(l) = *reinterpret_cast<const v8bf*>(g);
#endif
}

__device__ __forceinline__ void wait_async() {
#if ASYNC_LDS
#if __has_builtin(__builtin_amdgcn_s_wait_asynccnt)
    __builtin_amdgcn_s_wait_asynccnt(0);
#else
    asm volatile("s_wait_asynccnt 0" ::: "memory");
#endif
#endif
}

// Load a 16x32 bf16 fragment (rows x K, row-major, ld=32) in the ISA A/B VGPR
// swizzle: lane L<16 -> row L, K 0..7 then 16..23; lane L+16 -> row L, K 8..15 then 24..31.
__device__ __forceinline__ v16bf load_frag(const __bf16* tile, int lane) {
    const __bf16* p = tile + (lane & 15) * 32 + ((lane >> 4) << 3);
    v8bf lo = *reinterpret_cast<const v8bf*>(p);
    v8bf hi = *reinterpret_cast<const v8bf*>(p + 16);
    v16bf f;
#pragma unroll
    for (int i = 0; i < 8; ++i) { f[i] = lo[i]; f[i + 8] = hi[i]; }
    return f;
}

// Stage one 32-wide K chunk (A tile 64x32, B tile 128x32) into LDS.
__device__ __forceinline__ void stage_chunk(
    int j, int n1,
    const __bf16* __restrict__ A1, int K1, const __bf16* __restrict__ W1,
    const __bf16* __restrict__ A2, int K2, const __bf16* __restrict__ W2,
    int m0, int n0, int tid, __bf16* a_dst, __bf16* b_dst)
{
    const __bf16* A; const __bf16* W; int K, k0;
    if (j < n1) { A = A1; W = W1; K = K1; k0 = j * 32; }
    else        { A = A2; W = W2; K = K2; k0 = (j - n1) * 32; }
    {   // A: 8 bf16 (16B) per thread
        const int idx = tid << 3;
        const int r = idx >> 5, c = idx & 31;
        copy16(&A[(m0 + r) * K + k0 + c], &a_dst[idx]);
    }
    {   // B: 16 bf16 (2x16B) per thread
        const int r = tid >> 1;
        const int c = (tid & 1) << 4;
        const __bf16* src = &W[(n0 + r) * K + k0 + c];
        copy16(src,     &b_dst[r * 32 + c]);
        copy16(src + 8, &b_dst[r * 32 + c + 8]);
        if (k0 + 32 < K) __builtin_prefetch(src + 32, 0, 1);
    }
}

// C[256,N] = A1[256,K1] @ W1[N,K1]^T (+ A2[256,K2] @ W2[N,K2]^T) + bias[N]
// One fused K-pipeline over the concatenated reduction; double-buffered LDS,
// async global->LDS staging overlapped with WMMA. Block tile 64x128, 8 waves,
// wave tile 16x64 (4 x v_wmma_f32_16x16x32_bf16 per chunk).
__global__ __launch_bounds__(256) void gemm_dual(
    const __bf16* __restrict__ A1, int K1, const __bf16* __restrict__ W1,
    const __bf16* __restrict__ A2, int K2, const __bf16* __restrict__ W2,
    const float* __restrict__ bias, float* __restrict__ C, int N)
{
    __shared__ __align__(16) __bf16 lA[2][64 * 32];
    __shared__ __align__(16) __bf16 lB[2][128 * 32];
    const int tid  = threadIdx.x;
    const int lane = tid & 31;
    const int wave = tid >> 5;
    const int wm   = wave & 3;     // 4 waves along M
    const int wn   = wave >> 2;    // 2 waves along N
    const int m0   = blockIdx.y * 64;
    const int n0   = blockIdx.x * 128;

    const int n1 = K1 >> 5;
    const int n2 = K2 >> 5;
    const int nt = n1 + n2;

    v8f acc[4] = {};

    // prologue: stage chunk 0 into buffer 0
    stage_chunk(0, n1, A1, K1, W1, A2, K2, W2, m0, n0, tid, lA[0], lB[0]);

    for (int i = 0; i < nt; ++i) {
        wait_async();          // my transfers into buf (i&1) are complete
        __syncthreads();       // everyone's are; everyone done reading buf ((i+1)&1)
        if (i + 1 < nt)
            stage_chunk(i + 1, n1, A1, K1, W1, A2, K2, W2, m0, n0, tid,
                        lA[(i + 1) & 1], lB[(i + 1) & 1]);
        const __bf16* aT = lA[i & 1];
        const __bf16* bT = lB[i & 1];
        // hoist ALL fragment loads ahead of the WMMAs: one DS clause, then
        // back-to-back matrix ops (waits can drain incrementally).
        const v16bf af = load_frag(&aT[(wm * 16) * 32], lane);
        v16bf bfr[4];
#pragma unroll
        for (int j = 0; j < 4; ++j)
            bfr[j] = load_frag(&bT[(wn * 64 + j * 16) * 32], lane);
#pragma unroll
        for (int j = 0; j < 4; ++j)
            acc[j] = __builtin_amdgcn_wmma_f32_16x16x32_bf16(
                false, af, false, bfr[j], (short)0, acc[j], false, false);
    }

    // C/D layout: VGPR v -> M = v + 8*(lane>=16), N = lane & 15
    const int mh = (lane >> 4) << 3;
    const int nc = lane & 15;
#pragma unroll
    for (int j = 0; j < 4; ++j) {
        const int col = n0 + wn * 64 + j * 16 + nc;
        const float bv = bias[col];
#pragma unroll
        for (int v = 0; v < 8; ++v) {
            const int row = m0 + wm * 16 + mh + v;
            C[row * N + col] = acc[j][v] + bv;
        }
    }
}

__device__ __forceinline__ float sigm(float x) { return 1.0f / (1.0f + expf(-x)); }

// gates (B,4H) i|f|g|o ; c updated in place; h written as bf16 for next GEMMs
__global__ void lstm_pointwise(const float* __restrict__ gates, float* __restrict__ c,
                               __bf16* __restrict__ h_bf)
{
    const int idx = blockIdx.x * blockDim.x + threadIdx.x;   // Bb*Hh
    const int b = idx >> 10, j = idx & (Hh - 1);
    const float* g = gates + b * G4H;
    const float i_ = sigm(g[j]);
    const float f_ = sigm(g[Hh + j]);
    const float g_ = tanhf(g[2 * Hh + j]);
    const float o_ = sigm(g[3 * Hh + j]);
    const float cy = c[idx] * f_ + i_ * g_;
    c[idx] = cy;
    h_bf[idx] = f2bf(o_ * tanhf(cy));
}

// inp = mask_prev * x_t + (1-mask_prev) * xhat0_prev  -> bf16
__global__ void inp_stage(const float* __restrict__ x, int t,
                          const float* __restrict__ mask_prev,
                          const float* __restrict__ xhat0_prev,
                          __bf16* __restrict__ inp_bf)
{
    const int idx = blockIdx.x * blockDim.x + threadIdx.x;   // Bb*Dd
    const int b = idx >> 7, d = idx & (Dd - 1);
    const float xv = x[(b * Tt + t) * Dd + d];
    const float mp = mask_prev[idx];
    inp_bf[idx] = f2bf(mp * xv + (1.0f - mp) * xhat0_prev[idx]);
}

// mu -> mask/reg/mu_det/prob_det/z outputs, carry mask, inp2 = mask * x_{t+1}
__global__ void mask_stage(const float* __restrict__ mu, const float* __restrict__ x,
                           int t, float* __restrict__ out,
                           float* __restrict__ mask_ws, __bf16* __restrict__ inp2_bf)
{
    const int idx = blockIdx.x * blockDim.x + threadIdx.x;   // Bb*Dd
    const int b = idx >> 7, d = idx & (Dd - 1);
    const float m  = mu[idx];
    const float mk = fminf(fmaxf(m + 0.5f, 0.0f), 1.0f);
    const float rg = 0.5f * (1.0f + erff((m + 0.5f) * 1.41421356237f)); // /(0.5*sqrt2)
    const size_t slice = (size_t)Bb * TT * Dd;
    const size_t pos = ((size_t)b * TT + t) * Dd + d;
    out[0 * slice + pos] = mk;   // mask
    out[3 * slice + pos] = rg;   // reg
    out[4 * slice + pos] = m;    // mu_det
    out[5 * slice + pos] = rg;   // prob_det
    out[6 * slice + pos] = m;    // z
    mask_ws[idx] = mk;
    const float xv = x[(b * Tt + (t + 1)) * Dd + d];
    inp2_bf[idx] = f2bf(mk * xv);
}

// xhat (B,2D) -> out xhat0/xhat1 slices, carry xhat0
__global__ void fc_post(const float* __restrict__ xhat, int t, float* __restrict__ out,
                        float* __restrict__ xhat0_ws)
{
    const int idx = blockIdx.x * blockDim.x + threadIdx.x;   // Bb*Dd
    const int b = idx >> 7, d = idx & (Dd - 1);
    const float x0 = xhat[b * 2 * Dd + d];
    const float x1 = xhat[b * 2 * Dd + Dd + d];
    const size_t slice = (size_t)Bb * TT * Dd;
    const size_t pos = ((size_t)b * TT + t) * Dd + d;
    out[1 * slice + pos] = x0;
    out[2 * slice + pos] = x1;
    xhat0_ws[idx] = x0;
}

__global__ void cvt_bf16(const float* __restrict__ s, __bf16* __restrict__ d, int n) {
    const int i = blockIdx.x * blockDim.x + threadIdx.x;
    if (i < n) d[i] = f2bf(s[i]);
}
__global__ void vec_add(const float* __restrict__ a, const float* __restrict__ b,
                        float* __restrict__ o, int n) {
    const int i = blockIdx.x * blockDim.x + threadIdx.x;
    if (i < n) o[i] = a[i] + b[i];
}
__global__ void fill_f32(float* p, float v, int n) {
    const int i = blockIdx.x * blockDim.x + threadIdx.x;
    if (i < n) p[i] = v;
}
__global__ void fill_u16(unsigned short* p, int n) {
    const int i = blockIdx.x * blockDim.x + threadIdx.x;
    if (i < n) p[i] = 0;
}

extern "C" void kernel_launch(void* const* d_in, const int* in_sizes, int n_in,
                              void* d_out, int out_size, void* d_ws, size_t ws_size,
                              hipStream_t stream)
{
    const float* x       = (const float*)d_in[0];
    const float* s_x2h_w = (const float*)d_in[1];
    const float* s_x2h_b = (const float*)d_in[2];
    const float* s_h2h_w = (const float*)d_in[3];
    const float* s_h2h_b = (const float*)d_in[4];
    const float* p_x2h_w = (const float*)d_in[5];
    const float* p_x2h_b = (const float*)d_in[6];
    const float* p_h2h_w = (const float*)d_in[7];
    const float* p_h2h_b = (const float*)d_in[8];
    const float* p_fc_w  = (const float*)d_in[9];
    const float* p_fc_b  = (const float*)d_in[10];
    const float* fs_w    = (const float*)d_in[11];
    const float* fs_b    = (const float*)d_in[12];
    float* out = (float*)d_out;
    (void)in_sizes; (void)n_in; (void)out_size; (void)ws_size;

    // ---- workspace carve-out (~28 MB) ----
    char* w = (char*)d_ws;
    auto carve = [&](size_t bytes) { char* p = w; w += (bytes + 255) & ~(size_t)255; return p; };
    __bf16* s_x2h_bf = (__bf16*)carve((size_t)G4H * Dd * 2);
    __bf16* s_h2h_bf = (__bf16*)carve((size_t)G4H * Hh * 2);
    __bf16* p_x2h_bf = (__bf16*)carve((size_t)G4H * Dd * 2);
    __bf16* p_h2h_bf = (__bf16*)carve((size_t)G4H * Hh * 2);
    __bf16* p_fc_bf  = (__bf16*)carve((size_t)2 * Dd * Hh * 2);
    __bf16* fs_bf    = (__bf16*)carve((size_t)Dd * Hh * 2);
    float* s_bias    = (float*)carve((size_t)G4H * 4);
    float* p_bias    = (float*)carve((size_t)G4H * 4);
    float* gates     = (float*)carve((size_t)Bb * G4H * 4);
    float* mu        = (float*)carve((size_t)Bb * Dd * 4);
    float* xhat      = (float*)carve((size_t)Bb * 2 * Dd * 4);
    float* cn        = (float*)carve((size_t)Bb * Hh * 4);
    float* cn2       = (float*)carve((size_t)Bb * Hh * 4);
    __bf16* hn_bf    = (__bf16*)carve((size_t)Bb * Hh * 2);
    __bf16* hn2_bf   = (__bf16*)carve((size_t)Bb * Hh * 2);
    float* mask_ws   = (float*)carve((size_t)Bb * Dd * 4);
    float* xhat0_ws  = (float*)carve((size_t)Bb * Dd * 4);
    __bf16* inp_bf   = (__bf16*)carve((size_t)Bb * Dd * 2);
    __bf16* inp2_bf  = (__bf16*)carve((size_t)Bb * Dd * 2);

    const int TB = 256;
    auto blocks = [](int n) { return (n + 255) / 256; };

    // ---- one-time prep: bf16 weights, fused biases, zero/one state ----
    cvt_bf16<<<blocks(G4H * Dd), TB, 0, stream>>>(s_x2h_w, s_x2h_bf, G4H * Dd);
    cvt_bf16<<<blocks(G4H * Hh), TB, 0, stream>>>(s_h2h_w, s_h2h_bf, G4H * Hh);
    cvt_bf16<<<blocks(G4H * Dd), TB, 0, stream>>>(p_x2h_w, p_x2h_bf, G4H * Dd);
    cvt_bf16<<<blocks(G4H * Hh), TB, 0, stream>>>(p_h2h_w, p_h2h_bf, G4H * Hh);
    cvt_bf16<<<blocks(2 * Dd * Hh), TB, 0, stream>>>(p_fc_w, p_fc_bf, 2 * Dd * Hh);
    cvt_bf16<<<blocks(Dd * Hh), TB, 0, stream>>>(fs_w, fs_bf, Dd * Hh);
    vec_add<<<blocks(G4H), TB, 0, stream>>>(s_x2h_b, s_h2h_b, s_bias, G4H);
    vec_add<<<blocks(G4H), TB, 0, stream>>>(p_x2h_b, p_h2h_b, p_bias, G4H);
    fill_f32<<<blocks(Bb * Dd), TB, 0, stream>>>(mask_ws, 1.0f, Bb * Dd);
    fill_f32<<<blocks(Bb * Dd), TB, 0, stream>>>(xhat0_ws, 0.0f, Bb * Dd);
    fill_f32<<<blocks(Bb * Hh), TB, 0, stream>>>(cn, 0.0f, Bb * Hh);
    fill_f32<<<blocks(Bb * Hh), TB, 0, stream>>>(cn2, 0.0f, Bb * Hh);
    fill_u16<<<blocks(Bb * Hh), TB, 0, stream>>>((unsigned short*)hn_bf, Bb * Hh);
    fill_u16<<<blocks(Bb * Hh), TB, 0, stream>>>((unsigned short*)hn2_bf, Bb * Hh);

    const dim3 gGates(G4H / 128, Bb / 64);
    const dim3 gMu(Dd / 128, Bb / 64);
    const dim3 gFc(2 * Dd / 128, Bb / 64);

    // ---- sequential scan: 127 steps, 9 launches/step ----
    for (int t = 0; t < TT; ++t) {
        inp_stage<<<blocks(Bb * Dd), TB, 0, stream>>>(x, t, mask_ws, xhat0_ws, inp_bf);
        gemm_dual<<<gGates, TB, 0, stream>>>(inp_bf, Dd, s_x2h_bf,
                                             hn_bf, Hh, s_h2h_bf, s_bias, gates, G4H);
        lstm_pointwise<<<blocks(Bb * Hh), TB, 0, stream>>>(gates, cn, hn_bf);
        gemm_dual<<<gMu, TB, 0, stream>>>(hn_bf, Hh, fs_bf,
                                          nullptr, 0, nullptr, fs_b, mu, Dd);
        mask_stage<<<blocks(Bb * Dd), TB, 0, stream>>>(mu, x, t, out, mask_ws, inp2_bf);
        gemm_dual<<<gGates, TB, 0, stream>>>(inp2_bf, Dd, p_x2h_bf,
                                             hn2_bf, Hh, p_h2h_bf, p_bias, gates, G4H);
        lstm_pointwise<<<blocks(Bb * Hh), TB, 0, stream>>>(gates, cn2, hn2_bf);
        gemm_dual<<<gFc, TB, 0, stream>>>(hn2_bf, Hh, p_fc_bf,
                                          nullptr, 0, nullptr, p_fc_b, xhat, 2 * Dd);
        fc_post<<<blocks(Bb * Dd), TB, 0, stream>>>(xhat, t, out, xhat0_ws);
    }
}
